// TSARLayer_32727650796180
// MI455X (gfx1250) — matrix operand order repaired and checked by reading, not compiled, not run on hardware
//
#include <hip/hip_runtime.h>
#include <hip/hip_bf16.h>

// ---------------------------------------------------------------------------
// TSAR graph layer for MI455X (gfx1250, wave32, WMMA)
//
//   msg  = relu([feat[src] | eattr | etime] @ Wmsg^T + bmsg)   [E,192]x[192,128]
//   acc  = boundary + segment_sum(msg by dst)                  (f32 atomics)
//   out  = relu(LN(acc @ Wlin^T + blin))                       [N,128]x[128,128]
//
// Matrix math runs on v_wmma_f32_16x16x32_bf16 (fp32 accumulate). Weights are
// pre-converted once per launch to bf16 [K][N] layout in d_ws so B fragments
// load as 32 contiguous bytes per k row across 16 lanes. A fragments are built in LDS in the ISA
// fragment layout. The edge kernel double-buffers the A tile in LDS: tile t+1's
// global gather (src indices preloaded one tile further ahead) is issued before
// tile t's WMMA+scatter, so the L2 round-trip hides behind matrix math.
// ---------------------------------------------------------------------------

typedef __attribute__((ext_vector_type(16))) __bf16 v16bf;
typedef __attribute__((ext_vector_type(4)))  __bf16 v4bf;
typedef __attribute__((ext_vector_type(8)))  float  v8f;

#define N_NODES 50000
#define N_EDGES 800000
#define EMB     128
#define MSG_IN  192           // EMB + EA(32) + ET(32)
#define LN_EPS  1e-5f
#define EDGE_TILES_PER_BLOCK 8

// -------------------- kernel 0: weight convert/transpose --------------------
// Wb[k*128 + n] = (bf16) W[n*K + k]   for both weight matrices.
__global__ void __launch_bounds__(256)
prep_weights_kernel(const float* __restrict__ Wmsg, const float* __restrict__ Wlin,
                    __bf16* __restrict__ wbMsg, __bf16* __restrict__ wbLin) {
    int i = blockIdx.x * 256 + threadIdx.x;          // grid covers 24576+16384
    if (i < MSG_IN * EMB) {
        int k = i >> 7, n = i & 127;
        wbMsg[i] = (__bf16)Wmsg[n * MSG_IN + k];
    } else {
        int j = i - MSG_IN * EMB;
        int k = j >> 7, n = j & 127;
        wbLin[j] = (__bf16)Wlin[n * EMB + k];
    }
}

// -------------------- kernel 1: accumulator = boundary ----------------------
__global__ void __launch_bounds__(256)
init_acc_kernel(const float4* __restrict__ bc, float4* __restrict__ acc) {
    int i = blockIdx.x * 256 + threadIdx.x;          // N*EMB/4 elements
    acc[i] = bc[i];
}

// -------------------- kernel 2: edge message MLP + scatter ------------------
__global__ void __launch_bounds__(256)
edge_msg_kernel(const float* __restrict__ feat,
                const float* __restrict__ eattr,
                const float* __restrict__ etime,
                const int*   __restrict__ eidx,    // [2,E]: row0=src, row1=dst
                const __bf16* __restrict__ wb,     // [192][128] bf16
                const float* __restrict__ bmsg,
                float* __restrict__ acc) {
    // Double-buffered A fragments (6 k-chunks, ISA 16-bit A 16x32 layout).
    __shared__ __align__(32) __bf16 sA[2][6 * 32 * 16];

    const int t    = threadIdx.x;
    const int lane = t & 31;
    const int w    = t >> 5;                 // wave id = N-tile (0..7)

    // B fragments: lane%16 = column w*16+n, element s = K row in table order.
    v16bf bfrag[6];
#pragma unroll
    for (int kc = 0; kc < 6; ++kc)
#pragma unroll
        for (int s = 0; s < 16; ++s)
            bfrag[kc][s] = wb[(kc * 32 + ((s >> 3) << 4) + ((lane >> 4) << 3) + (s & 7)) * EMB + w * 16 + (lane & 15)];

    const float bias = bmsg[w * 16 + (lane & 15)];
    const int   mhi  = (lane >> 4) << 3;     // C/D: rows 0..7 or 8..15
    const int   ncol = w * 16 + (lane & 15);

    // Per-thread gather geometry: 3 aligned float4 quads of the 16x192 tile.
    int rows[3], cols[3], lds[3];
#pragma unroll
    for (int j = 0; j < 3; ++j) {
        int q   = j * 256 + t;               // 0..767 quads (16 rows x 48)
        int row = q / 48;
        int col = (q - row * 48) << 2;       // multiple of 4
        rows[j] = row;
        cols[j] = col;
        int kc = col >> 5, k = col & 31;
        int l  = row + ((k & 8)  ? 16 : 0);
        int el = ((k & 16) ? 8 : 0) + (k & 7);
        lds[j] = (kc * 32 + l) * 16 + el;
    }

    const int tile0 = blockIdx.x * EDGE_TILES_PER_BLOCK;
    int    sidxN[3], sidxNN[3];
    float4 val[3];

    // ---- prologue: src indices + data for tile 0, src indices for tile 1 ---
#pragma unroll
    for (int j = 0; j < 3; ++j)
        sidxN[j] = (cols[j] < EMB) ? eidx[tile0 * 16 + rows[j]] : 0;
#pragma unroll
    for (int j = 0; j < 3; ++j) {
        int e = tile0 * 16 + rows[j];
        const float* p = (cols[j] < EMB)      ? &feat[sidxN[j] * EMB + cols[j]]
                       : (cols[j] < EMB + 32) ? &eattr[e * 32 + (cols[j] - EMB)]
                                              : &etime[e * 32 + (cols[j] - EMB - 32)];
        val[j] = *(const float4*)p;
    }
#pragma unroll
    for (int j = 0; j < 3; ++j)
        sidxN[j] = (cols[j] < EMB) ? eidx[(tile0 + 1) * 16 + rows[j]] : 0;
#pragma unroll
    for (int j = 0; j < 3; ++j) {
        v4bf b;
        b.x = (__bf16)val[j].x; b.y = (__bf16)val[j].y;
        b.z = (__bf16)val[j].z; b.w = (__bf16)val[j].w;
        *(v4bf*)&sA[0][lds[j]] = b;
    }
    __syncthreads();

    // ---- pipelined main loop ----
    for (int et = 0; et < EDGE_TILES_PER_BLOCK; ++et) {
        const int ebase = (tile0 + et) * 16;
        const int cur   = et & 1;
        const bool more = (et + 1 < EDGE_TILES_PER_BLOCK);

        // Issue next tile's gather early; waits land after the WMMA block.
        if (more) {
            const int enext = ebase + 16;
#pragma unroll
            for (int j = 0; j < 3; ++j) {
                int e = enext + rows[j];
                const float* p = (cols[j] < EMB)      ? &feat[sidxN[j] * EMB + cols[j]]
                               : (cols[j] < EMB + 32) ? &eattr[e * 32 + (cols[j] - EMB)]
                                                      : &etime[e * 32 + (cols[j] - EMB - 32)];
                val[j] = *(const float4*)p;
            }
            // src indices two tiles ahead (in-bounds: reads stay inside eidx[2*E])
#pragma unroll
            for (int j = 0; j < 3; ++j)
                sidxNN[j] = (cols[j] < EMB) ? eidx[enext + 16 + rows[j]] : 0;
        }

        // 16x16x192 bf16 WMMA chain for this wave's N-tile.
        v8f d = { bias, bias, bias, bias, bias, bias, bias, bias };
#pragma unroll
        for (int kc = 0; kc < 6; ++kc) {
            v16bf a = *(const v16bf*)&sA[cur][(kc * 32 + lane) * 16];
            d = __builtin_amdgcn_wmma_f32_16x16x32_bf16(
                    false, a, false, bfrag[kc], (short)0, d, false, false);
        }

        // ReLU + scatter-add into node accumulator (L2-resident, 25.6 MB).
#pragma unroll
        for (int r = 0; r < 8; ++r) {
            float v = fmaxf(d[r], 0.0f);
            int   e = ebase + r + mhi;
            int  dn = eidx[N_EDGES + e];     // dst node
            unsafeAtomicAdd(&acc[dn * EMB + ncol], v);
        }

        // Convert+store next tile into the other buffer; rotate index pipeline.
        if (more) {
#pragma unroll
            for (int j = 0; j < 3; ++j) {
                v4bf b;
                b.x = (__bf16)val[j].x; b.y = (__bf16)val[j].y;
                b.z = (__bf16)val[j].z; b.w = (__bf16)val[j].w;
                *(v4bf*)&sA[1 - cur][lds[j]] = b;
            }
#pragma unroll
            for (int j = 0; j < 3; ++j) sidxN[j] = sidxNN[j];
            __syncthreads();
        }
    }
}

// -------------------- kernel 3: node linear + LayerNorm + ReLU --------------
__global__ void __launch_bounds__(256)
node_update_kernel(float* __restrict__ out,          // acc in, final out
                   const __bf16* __restrict__ wb,    // [128][128] bf16
                   const float* __restrict__ blin,
                   const float* __restrict__ gamma,
                   const float* __restrict__ beta) {
    __shared__ __align__(32) __bf16 sA[4 * 32 * 16];
    __shared__ float sTile[16 * EMB];
    __shared__ float sMu[16], sRs[16];

    const int t    = threadIdx.x;
    const int lane = t & 31;
    const int w    = t >> 5;
    const int nbase = blockIdx.x * 16;       // this block owns 16 node rows

    v16bf bfrag[4];
#pragma unroll
    for (int kc = 0; kc < 4; ++kc)
#pragma unroll
        for (int s = 0; s < 16; ++s)
            bfrag[kc][s] = wb[(kc * 32 + ((s >> 3) << 4) + ((lane >> 4) << 3) + (s & 7)) * EMB + w * 16 + (lane & 15)];

    const float bias = blin[w * 16 + (lane & 15)];

    // Read this block's 16 accumulated rows (float4 quads) into bf16 fragments.
    float4 v[2];
    int ldsoff[2];
#pragma unroll
    for (int it = 0; it < 2; ++it) {
        int q   = it * 256 + t;              // 0..511 quads (16 rows x 32)
        int row = q >> 5;
        int col = (q & 31) << 2;
        v[it] = *(const float4*)&out[(nbase + row) * EMB + col];
        int kc = col >> 5, k = col & 31;
        int l  = row + ((k & 8)  ? 16 : 0);
        int el = ((k & 16) ? 8 : 0) + (k & 7);
        ldsoff[it] = (kc * 32 + l) * 16 + el;
    }
#pragma unroll
    for (int it = 0; it < 2; ++it) {
        v4bf b;
        b.x = (__bf16)v[it].x; b.y = (__bf16)v[it].y;
        b.z = (__bf16)v[it].z; b.w = (__bf16)v[it].w;
        *(v4bf*)&sA[ldsoff[it]] = b;
    }
    __syncthreads();

    v8f d = { bias, bias, bias, bias, bias, bias, bias, bias };
#pragma unroll
    for (int kc = 0; kc < 4; ++kc) {
        v16bf a = *(const v16bf*)&sA[(kc * 32 + lane) * 16];
        d = __builtin_amdgcn_wmma_f32_16x16x32_bf16(
                false, a, false, bfrag[kc], (short)0, d, false, false);
    }

    // Stage 16x128 result in LDS for the cross-wave LayerNorm.
    const int mhi  = (lane >> 4) << 3;
    const int ncol = w * 16 + (lane & 15);
#pragma unroll
    for (int r = 0; r < 8; ++r)
        sTile[(r + mhi) * EMB + ncol] = d[r];
    __syncthreads();

    if (t < 16) {
        float s = 0.0f, s2 = 0.0f;
        for (int c = 0; c < EMB; ++c) {
            float x = sTile[t * EMB + c];
            s += x; s2 += x * x;
        }
        float mu  = s * (1.0f / EMB);
        float var = s2 * (1.0f / EMB) - mu * mu;
        sMu[t] = mu;
        sRs[t] = rsqrtf(var + LN_EPS);
    }
    __syncthreads();

#pragma unroll
    for (int it = 0; it < 8; ++it) {
        int i   = it * 256 + t;
        int row = i >> 7;
        int col = i & 127;
        float x = (sTile[i] - sMu[row]) * sRs[row] * gamma[col] + beta[col];
        out[nbase * EMB + i] = fmaxf(x, 0.0f);
    }
}

// ---------------------------------------------------------------------------
extern "C" void kernel_launch(void* const* d_in, const int* in_sizes, int n_in,
                              void* d_out, int out_size, void* d_ws, size_t ws_size,
                              hipStream_t stream) {
    const float* feat  = (const float*)d_in[0];
    const float* eattr = (const float*)d_in[1];
    const float* etime = (const float*)d_in[2];
    const float* bc    = (const float*)d_in[3];
    const int*   eidx  = (const int*)  d_in[4];
    const float* Wmsg  = (const float*)d_in[5];
    const float* bmsg  = (const float*)d_in[6];
    const float* Wlin  = (const float*)d_in[7];
    const float* blin  = (const float*)d_in[8];
    const float* g     = (const float*)d_in[9];
    const float* bt    = (const float*)d_in[10];
    float* out = (float*)d_out;

    // Workspace: bf16 weights only (~80 KB). Accumulation happens in d_out.
    __bf16* wbMsg = (__bf16*)d_ws;
    __bf16* wbLin = (__bf16*)((char*)d_ws + (size_t)MSG_IN * EMB * sizeof(__bf16));

    prep_weights_kernel<<<(MSG_IN * EMB + EMB * EMB) / 256, 256, 0, stream>>>(
        Wmsg, Wlin, wbMsg, wbLin);
    init_acc_kernel<<<(N_NODES * EMB / 4) / 256, 256, 0, stream>>>(
        (const float4*)bc, (float4*)out);
    edge_msg_kernel<<<N_EDGES / (16 * EDGE_TILES_PER_BLOCK), 256, 0, stream>>>(
        feat, eattr, etime, eidx, wbMsg, bmsg, out);
    node_update_kernel<<<N_NODES / 16, 256, 0, stream>>>(out, wbLin, blin, g, bt);
}
